// TransformerBlock_4853313045252
// MI455X (gfx1250) — matrix operand order repaired and checked by reading, not compiled, not run on hardware
//
#include <hip/hip_runtime.h>
#include <hip/hip_bf16.h>
#include <math.h>

typedef __bf16 bf16;
typedef __attribute__((ext_vector_type(16))) __bf16 v16bf;
typedef __attribute__((ext_vector_type(8)))  float  v8f;
typedef __attribute__((ext_vector_type(8)))  unsigned v8u;
typedef __attribute__((ext_vector_type(4)))  int v4i;
typedef __attribute__((ext_vector_type(2)))  int v2i;

#define NPOS  2304   // H*W
#define HH    48
#define WW    48
#define CDIM  192
#define NHEADS 6
#define HD    32
#define BATCH 2

struct __align__(16) B128 { unsigned long long x, y; };
struct __align__(8)  B64v { unsigned long long x; };

// ---------- async global->LDS staging (CDNA5), probe-gated with fallback ----------
#if __has_builtin(__builtin_amdgcn_global_load_async_to_lds_b128) && \
    __has_builtin(__builtin_amdgcn_global_load_async_to_lds_b64)  && \
    __has_builtin(__builtin_amdgcn_s_wait_asynccnt)
#define HAVE_ASYNC 1
#endif

__device__ __forceinline__ void async_b128(const void* g, void* l) {
#ifdef HAVE_ASYNC
  auto gp = (__attribute__((address_space(1))) v4i*)(unsigned long long)(uintptr_t)g;
  auto lp = (__attribute__((address_space(3))) v4i*)(unsigned)(uintptr_t)l;
  __builtin_amdgcn_global_load_async_to_lds_b128(gp, lp, 0, 0);
#else
  *(B128*)l = *(const B128*)g;
#endif
}
__device__ __forceinline__ void async_b64(const void* g, void* l) {
#ifdef HAVE_ASYNC
  auto gp = (__attribute__((address_space(1))) v2i*)(unsigned long long)(uintptr_t)g;
  auto lp = (__attribute__((address_space(3))) v2i*)(unsigned)(uintptr_t)l;
  __builtin_amdgcn_global_load_async_to_lds_b64(gp, lp, 0, 0);
#else
  *(B64v*)l = *(const B64v*)g;
#endif
}
__device__ __forceinline__ void wait_async() {
#ifdef HAVE_ASYNC
  __builtin_amdgcn_s_wait_asynccnt(0);
#endif
}

__device__ __forceinline__ v16bf as_bf(v8u u) { return __builtin_bit_cast(v16bf, u); }

// ---------------- weight conversion (with padding) ----------------
__global__ void cvt_pad_kernel(const float* __restrict__ w, bf16* __restrict__ o,
                               int rows_out, int cols_out, int rows_real, int cols_real) {
  int i = blockIdx.x * blockDim.x + threadIdx.x;
  int tot = rows_out * cols_out;
  if (i >= tot) return;
  int r = i / cols_out, c = i % cols_out;
  float v = (r < rows_real && c < cols_real) ? w[r * cols_real + c] : 0.f;
  o[i] = (bf16)v;
}

// ---------------- LayerNorm over spatial dims, per (b,c) ----------------
__global__ __launch_bounds__(256) void ln2d_kernel(const float* __restrict__ x,
                                                   const float* __restrict__ w,
                                                   const float* __restrict__ b,
                                                   bf16* __restrict__ y, int C) {
  int bc = blockIdx.x;
  int c = bc % C;
  const float* xp = x + (size_t)bc * NPOS;
  bf16* yp = y + (size_t)bc * NPOS;
  __shared__ float s1[256], s2[256];
  float a = 0.f, q = 0.f;
  for (int i = threadIdx.x; i < NPOS; i += 256) { float v = xp[i]; a += v; q += v * v; }
  s1[threadIdx.x] = a; s2[threadIdx.x] = q; __syncthreads();
  for (int st = 128; st > 0; st >>= 1) {
    if (threadIdx.x < st) { s1[threadIdx.x] += s1[threadIdx.x + st]; s2[threadIdx.x] += s2[threadIdx.x + st]; }
    __syncthreads();
  }
  float m   = s1[0] / (float)NPOS;
  float var = s2[0] / (float)NPOS - m * m;
  float inv = rsqrtf(var + 1e-6f);
  float sw = w[c] * inv;
  float sb = b[c] - m * inv * w[c];
  for (int i = threadIdx.x; i < NPOS; i += 256) yp[i] = (bf16)(xp[i] * sw + sb);
}

// ---------------- generic bf16 WMMA GEMM: out[M,N] = W[M,K]*X[K,N] (+bias,+res) ----------------
__global__ __launch_bounds__(256) void gemm_bf16_kernel(
    const bf16* __restrict__ W, const bf16* __restrict__ X,
    const float* __restrict__ bias, const float* __restrict__ res,
    float* __restrict__ out, int M, int K, int Mreal) {
  const int N = NPOS;
  int bb = blockIdx.z;
  const bf16*  Xb   = X + (size_t)bb * K * N;
  const float* resb = res ? res + (size_t)bb * M * N : nullptr;
  float*       outb = out + (size_t)bb * M * N;
  int m0 = blockIdx.y * 64, n0 = blockIdx.x * 64;

  __shared__ bf16     As[64][32];     // [m][k]         (async-staged)
  __shared__ unsigned Bs2[16][64];    // k-pair interleaved: u32 {k=2p, k=2p+1} per col n

  int tid = threadIdx.x;
  int wv = tid >> 5, lane = tid & 31, l16 = lane & 15, half = lane >> 4;
  int mw = (wv & 3) * 16, nw = (wv >> 2) * 32;

  v8f acc0 = {}; v8f acc1 = {};

  for (int k0 = 0; k0 < K; k0 += 32) {
    { // W tile: contiguous b128 per thread, async to LDS
      int m = tid >> 2, kk = (tid & 3) * 8;
      async_b128(&W[(size_t)(m0 + m) * K + k0 + kk], &As[m][kk]);
    }
    { // X tile: two b64 row loads, interleave k-pairs, one b128 LDS store
      int kp = tid >> 4, nq = (tid & 15) * 4;
      unsigned long long r0 = *(const unsigned long long*)&Xb[(size_t)(k0 + 2 * kp)     * N + n0 + nq];
      unsigned long long r1 = *(const unsigned long long*)&Xb[(size_t)(k0 + 2 * kp + 1) * N + n0 + nq];
      unsigned w0 = (unsigned)(r0 & 0xffffu)         | ((unsigned)(r1 & 0xffffu) << 16);
      unsigned w1 = (unsigned)((r0 >> 16) & 0xffffu) | ((unsigned)((r1 >> 16) & 0xffffu) << 16);
      unsigned w2 = (unsigned)((r0 >> 32) & 0xffffu) | ((unsigned)((r1 >> 32) & 0xffffu) << 16);
      unsigned w3 = (unsigned)(r0 >> 48)             | ((unsigned)(r1 >> 48) << 16);
      *(uint4*)&Bs2[kp][nq] = make_uint4(w0, w1, w2, w3);
    }
    wait_async();
    __syncthreads();

    // A fragment: pairs contiguous -> one b32 LDS read per VGPR
    v8u au;
    {
      int r = mw + l16;
      #pragma unroll
      for (int v = 0; v < 8; ++v) {
        int kk = 2 * v + 16 * half;
        au[v] = *(const unsigned*)&As[r][kk];
      }
    }
    // B fragments: k-pair p = v + 8*half, col n -> one b32 read per VGPR
    v8u bu0, bu1;
    #pragma unroll
    for (int v = 0; v < 8; ++v) {
      int p = v + 8 * half;
      bu0[v] = Bs2[p][nw + l16];
      bu1[v] = Bs2[p][nw + 16 + l16];
    }
    acc0 = __builtin_amdgcn_wmma_f32_16x16x32_bf16(false, as_bf(au), false, as_bf(bu0), (short)0, acc0, false, false);
    acc1 = __builtin_amdgcn_wmma_f32_16x16x32_bf16(false, as_bf(au), false, as_bf(bu1), (short)0, acc1, false, false);
    __syncthreads();
  }

  int nn = n0 + nw + l16;
  if (resb) {
    #pragma unroll
    for (int v = 0; v < 8; ++v) {
      int m = m0 + mw + v + 8 * half;
      float bia = bias[m < Mreal ? m : Mreal - 1];
      bia = (m < Mreal) ? bia : 0.f;
      size_t i0 = (size_t)m * N + nn;
      outb[i0]      = acc0[v] + bia + resb[i0];
      outb[i0 + 16] = acc1[v] + bia + resb[i0 + 16];
    }
  } else {
    #pragma unroll
    for (int v = 0; v < 8; ++v) {
      int m = m0 + mw + v + 8 * half;
      float bia = bias[m < Mreal ? m : Mreal - 1];
      bia = (m < Mreal) ? bia : 0.f;
      size_t i0 = (size_t)m * N + nn;
      outb[i0]      = acc0[v] + bia;
      outb[i0 + 16] = acc1[v] + bia;
    }
  }
}

// ---------------- depthwise 3x3 SAME ----------------
__global__ void dwconv3_kernel(const float* __restrict__ in, const float* __restrict__ w,
                               const float* __restrict__ bias, float* __restrict__ out,
                               int C, int bstride) {
  int i = blockIdx.x * blockDim.x + threadIdx.x;
  int total = BATCH * C * NPOS;
  if (i >= total) return;
  int n = i % NPOS;
  int c = (i / NPOS) % C;
  int b = i / (NPOS * C);
  int hy = n / WW, wx = n % WW;
  const float* ip = in + (size_t)b * bstride + (size_t)c * NPOS;
  float s = bias[c];
  #pragma unroll
  for (int ky = -1; ky <= 1; ++ky) {
    int y = hy + ky; if (y < 0 || y >= HH) continue;
    #pragma unroll
    for (int kx = -1; kx <= 1; ++kx) {
      int xx = wx + kx; if (xx < 0 || xx >= WW) continue;
      s += ip[y * WW + xx] * w[c * 9 + (ky + 1) * 3 + (kx + 1)];
    }
  }
  out[(size_t)b * bstride + (size_t)c * NPOS + n] = s;
}

// ---------------- q/k per-channel L2 norm + bf16 convert (v: plain convert) ----------------
__global__ __launch_bounds__(256) void qkv_post_kernel(const float* __restrict__ qkv,
                                                       bf16* __restrict__ o) {
  int bc = blockIdx.x;           // b*576 + ch
  int ch = bc % 576;
  const float* p = qkv + (size_t)bc * NPOS;
  bf16* op = o + (size_t)bc * NPOS;
  float scale = 1.f;
  __shared__ float s[256];
  if (ch < 384) {                // q or k channel
    float q = 0.f;
    for (int i = threadIdx.x; i < NPOS; i += 256) { float v = p[i]; q += v * v; }
    s[threadIdx.x] = q; __syncthreads();
    for (int st = 128; st > 0; st >>= 1) {
      if (threadIdx.x < st) s[threadIdx.x] += s[threadIdx.x + st];
      __syncthreads();
    }
    scale = 1.f / fmaxf(sqrtf(s[0]), 1e-12f);
  }
  for (int i = threadIdx.x; i < NPOS; i += 256) op[i] = (bf16)(p[i] * scale);
}

// ---------------- flash attention: per wave one 16-query strip, 32-key tiles ----------------
__global__ __launch_bounds__(256) void flash_attn_kernel(const bf16* __restrict__ qkv,
                                                         bf16* __restrict__ o) {
  const int N = NPOS;
  int bh = blockIdx.y;
  int b = bh / NHEADS, h = bh % NHEADS;
  const bf16* qb = qkv + (size_t)b * 576 * N + (size_t)(h * HD) * N;
  const bf16* kb = qb + (size_t)CDIM * N;
  const bf16* vb = qb + (size_t)(2 * CDIM) * N;
  bf16* ob = o + (size_t)b * CDIM * N + (size_t)(h * HD) * N;

  __shared__ unsigned ksu[16][32];   // d-pair interleaved K tile: u32 {d=2p, d=2p+1} per key
  __shared__ bf16     vs[32][32];    // V tile natural [d][key]  (async-staged)
  __shared__ bf16     ps[8][16][32]; // per-wave P staging [qrow][key]

  int tid = threadIdx.x, wv = tid >> 5, lane = tid & 31, l16 = lane & 15, half = lane >> 4;
  int nq0 = blockIdx.x * 128 + wv * 16;

  // Q fragment (A layout): d-pairs packed into u32 directly from global
  v8u aqu;
  {
    int n = nq0 + l16;
    #pragma unroll
    for (int v = 0; v < 8; ++v) {
      int d = 2 * v + 16 * half;
      unsigned lo = *(const unsigned short*)&qb[(size_t)d * N + n];
      unsigned hi = *(const unsigned short*)&qb[(size_t)(d + 1) * N + n];
      aqu[v] = lo | (hi << 16);
    }
  }

  float mst[8], lst[8];
  #pragma unroll
  for (int v = 0; v < 8; ++v) { mst[v] = -3.0e38f; lst[v] = 0.f; }
  v8f o0 = {}; v8f o1 = {};
  const float scale = 0.1767766952966369f;   // 32^-0.5

  for (int k0 = 0; k0 < N; k0 += 32) {
    { // V tile: async b64 per thread
      int d = tid >> 3, j = (tid & 7) * 4;
      async_b64(&vb[(size_t)d * N + k0 + j], &vs[d][j]);
    }
    { // K tile: interleave d-pairs -> u32 per key column
      int dp = tid >> 4, j = (tid & 15) * 2;
      unsigned r0 = *(const unsigned*)&kb[(size_t)(2 * dp)     * N + k0 + j];
      unsigned r1 = *(const unsigned*)&kb[(size_t)(2 * dp + 1) * N + k0 + j];
      unsigned w0 = (r0 & 0xffffu) | ((r1 & 0xffffu) << 16);
      unsigned w1 = (r0 >> 16)     | ((r1 >> 16) << 16);
      *(uint2*)&ksu[dp][j] = make_uint2(w0, w1);
    }
    wait_async();
    __syncthreads();

    // S = Qt K for 32 keys: two 16-wide B fragments, one b32 read per VGPR
    v8u bk0, bk1;
    #pragma unroll
    for (int v = 0; v < 8; ++v) {
      int p = v + 8 * half;
      bk0[v] = ksu[p][l16];
      bk1[v] = ksu[p][16 + l16];
    }
    v8f z = {};
    v8f s0 = __builtin_amdgcn_wmma_f32_16x16x32_bf16(false, as_bf(aqu), false, as_bf(bk0), (short)0, z, false, false);
    v8f s1 = __builtin_amdgcn_wmma_f32_16x16x32_bf16(false, as_bf(aqu), false, as_bf(bk1), (short)0, z, false, false);

    // online softmax (row = v + 8*half, spread over 16 lanes of this half)
    #pragma unroll
    for (int v = 0; v < 8; ++v) {
      float e0 = s0[v] * scale, e1 = s1[v] * scale;
      float r = fmaxf(e0, e1);
      #pragma unroll
      for (int m = 8; m > 0; m >>= 1) r = fmaxf(r, __shfl_xor(r, m, 16));
      float nm = fmaxf(mst[v], r);
      float al = __expf(mst[v] - nm);
      float p0 = __expf(e0 - nm), p1 = __expf(e1 - nm);
      float rs = p0 + p1;
      #pragma unroll
      for (int m = 8; m > 0; m >>= 1) rs += __shfl_xor(rs, m, 16);
      lst[v] = lst[v] * al + rs;
      mst[v] = nm;
      o0[v] *= al; o1[v] *= al;
      int row = v + 8 * half;
      ps[wv][row][l16]      = (bf16)p0;
      ps[wv][row][16 + l16] = (bf16)p1;
    }

    // P (A layout, pairs contiguous -> b32 reads); V (B layout, pairs contiguous in vs rows)
    v8u pau;
    #pragma unroll
    for (int v = 0; v < 8; ++v) {
      int kk = 2 * v + 16 * half;
      pau[v] = *(const unsigned*)&ps[wv][l16][kk];
    }
    v8u bv0, bv1;
    #pragma unroll
    for (int v = 0; v < 8; ++v) {
      int mm = 2 * v + 16 * half;
      bv0[v] = *(const unsigned*)&vs[l16][mm];
      bv1[v] = *(const unsigned*)&vs[16 + l16][mm];
    }
    o0 = __builtin_amdgcn_wmma_f32_16x16x32_bf16(false, as_bf(pau), false, as_bf(bv0), (short)0, o0, false, false);
    o1 = __builtin_amdgcn_wmma_f32_16x16x32_bf16(false, as_bf(pau), false, as_bf(bv1), (short)0, o1, false, false);
    __syncthreads();
  }

  // write Ot: element (qrow = v+8*half, d = frag*16 + lane%16), normalize by l
  #pragma unroll
  for (int v = 0; v < 8; ++v) {
    float inv = 1.f / lst[v];
    int n = nq0 + v + 8 * half;
    ob[(size_t)l16 * N + n]        = (bf16)(o0[v] * inv);
    ob[(size_t)(16 + l16) * N + n] = (bf16)(o1[v] * inv);
  }
}

// ---------------- gated GELU (exact erf), writes K-padded bf16 [B,512,N] ----------------
__global__ void gelu_gate_kernel(const float* __restrict__ fin, bf16* __restrict__ g) {
  int i = blockIdx.x * blockDim.x + threadIdx.x;
  int total = BATCH * 512 * NPOS;
  if (i >= total) return;
  int n = i % NPOS;
  int c = (i / NPOS) % 512;
  int b = i / (512 * NPOS);
  float r = 0.f;
  if (c < 510) {
    float y1 = fin[(size_t)b * 1024 * NPOS + (size_t)c * NPOS + n];
    float y2 = fin[(size_t)b * 1024 * NPOS + (size_t)(510 + c) * NPOS + n];
    r = 0.5f * y1 * (1.f + erff(y1 * 0.70710678118654752f)) * y2;
  }
  g[(size_t)b * 512 * NPOS + (size_t)c * NPOS + n] = (bf16)r;
}

// ---------------- host orchestration ----------------
extern "C" void kernel_launch(void* const* d_in, const int* in_sizes, int n_in,
                              void* d_out, int out_size, void* d_ws, size_t ws_size,
                              hipStream_t stream) {
  (void)in_sizes; (void)n_in; (void)out_size; (void)ws_size;
  const float* x         = (const float*)d_in[0];
  const float* norm1_w   = (const float*)d_in[1];
  const float* norm1_b   = (const float*)d_in[2];
  const float* qkv_w     = (const float*)d_in[3];
  const float* qkv_b     = (const float*)d_in[4];
  const float* attn_dw_w = (const float*)d_in[5];
  const float* attn_dw_b = (const float*)d_in[6];
  const float* proj_w    = (const float*)d_in[7];
  const float* proj_b    = (const float*)d_in[8];
  const float* norm2_w   = (const float*)d_in[9];
  const float* norm2_b   = (const float*)d_in[10];
  const float* ffn_in_w  = (const float*)d_in[11];
  const float* ffn_in_b  = (const float*)d_in[12];
  const float* ffn_dw_w  = (const float*)d_in[13];
  const float* ffn_dw_b  = (const float*)d_in[14];
  const float* ffn_out_w = (const float*)d_in[15];
  const float* ffn_out_b = (const float*)d_in[16];
  float* out = (float*)d_out;

  char* p = (char*)d_ws;
  auto alloc = [&](size_t bytes) -> char* {
    char* r = p; p += (bytes + 255) & ~(size_t)255; return r;
  };
  bf16*  w_qkv  = (bf16*)alloc((size_t)576 * 192 * 2);
  bf16*  w_proj = (bf16*)alloc((size_t)192 * 192 * 2);
  bf16*  w_fin  = (bf16*)alloc((size_t)1024 * 192 * 2);   // M padded 1020->1024
  bf16*  w_fout = (bf16*)alloc((size_t)192 * 512 * 2);    // K padded 510->512
  bf16*  y_bf   = (bf16*)alloc((size_t)BATCH * 192 * NPOS * 2);
  float* regA   = (float*)alloc((size_t)BATCH * 1024 * NPOS * 4);  // qkv gemm out / ffn1
  float* regB   = (float*)alloc((size_t)BATCH * 1024 * NPOS * 4);  // qkv dwconv / ffn2
  bf16*  qkv_bf = (bf16*)alloc((size_t)BATCH * 576 * NPOS * 2);
  bf16*  o_bf   = (bf16*)alloc((size_t)BATCH * 192 * NPOS * 2);
  float* x1     = (float*)alloc((size_t)BATCH * 192 * NPOS * 4);
  bf16*  g_bf   = (bf16*)alloc((size_t)BATCH * 512 * NPOS * 2);

  auto cdiv = [](int a, int b) { return (a + b - 1) / b; };

  // weight conversion
  cvt_pad_kernel<<<cdiv(576 * 192, 256), 256, 0, stream>>>(qkv_w, w_qkv, 576, 192, 576, 192);
  cvt_pad_kernel<<<cdiv(192 * 192, 256), 256, 0, stream>>>(proj_w, w_proj, 192, 192, 192, 192);
  cvt_pad_kernel<<<cdiv(1024 * 192, 256), 256, 0, stream>>>(ffn_in_w, w_fin, 1024, 192, 1020, 192);
  cvt_pad_kernel<<<cdiv(192 * 512, 256), 256, 0, stream>>>(ffn_out_w, w_fout, 192, 512, 192, 510);

  // --- attention branch ---
  ln2d_kernel<<<BATCH * 192, 256, 0, stream>>>(x, norm1_w, norm1_b, y_bf, 192);
  gemm_bf16_kernel<<<dim3(36, 576 / 64, BATCH), 256, 0, stream>>>(
      w_qkv, y_bf, qkv_b, nullptr, regA, 576, 192, 576);
  dwconv3_kernel<<<cdiv(BATCH * 576 * NPOS, 256), 256, 0, stream>>>(
      regA, attn_dw_w, attn_dw_b, regB, 576, 576 * NPOS);
  qkv_post_kernel<<<BATCH * 576, 256, 0, stream>>>(regB, qkv_bf);
  flash_attn_kernel<<<dim3(NPOS / 128, BATCH * NHEADS), 256, 0, stream>>>(qkv_bf, o_bf);
  gemm_bf16_kernel<<<dim3(36, 192 / 64, BATCH), 256, 0, stream>>>(
      w_proj, o_bf, proj_b, x, x1, 192, 192, 192);

  // --- gated depthwise FFN branch ---
  ln2d_kernel<<<BATCH * 192, 256, 0, stream>>>(x1, norm2_w, norm2_b, y_bf, 192);
  gemm_bf16_kernel<<<dim3(36, 1024 / 64, BATCH), 256, 0, stream>>>(
      w_fin, y_bf, ffn_in_b, nullptr, regA, 1024, 192, 1020);
  dwconv3_kernel<<<cdiv(BATCH * 1020 * NPOS, 256), 256, 0, stream>>>(
      regA, ffn_dw_w, ffn_dw_b, regB, 1020, 1024 * NPOS);
  gelu_gate_kernel<<<cdiv(BATCH * 512 * NPOS, 256), 256, 0, stream>>>(regB, g_bf);
  gemm_bf16_kernel<<<dim3(36, 192 / 64, BATCH), 256, 0, stream>>>(
      w_fout, g_bf, ffn_out_b, x1, out, 192, 512, 192);
}
